// MultiHeadAttention_50783693308091
// MI455X (gfx1250) — matrix operand order repaired and checked by reading, not compiled, not run on hardware
//
#include <hip/hip_runtime.h>

// ---------------------------------------------------------------------------
// MHA forward for MI455X (gfx1250, wave32, WMMA).
//   B=2, S=2048, H=1024, 16 heads, head_dim=64.
// All GEMMs via v_wmma_f32_16x16x32_f16 (f16 inputs, f32 accum).
// Reference uses a RAW reshape (b,i,H)->(b,n,i,d): each head's matrix is the
// contiguous 2048x64 block at per-batch flat offset n*S*HD, row-major, ld=HD.
// Workspace (halves): qf,kf,vf (3*4M) | wq,wk,wv,wo (4*1M)
//                     | Qp,Kp (2*4M) | Vt (4M, per-head transposed) | Xh (4M)
// Total 32M halves = 64 MB.
// Round-3 changes: explicit ping-pong double buffering in gemm_tile (kills the
// v_dual_mov_b32 rotation copies), V-fragment loads hoisted ahead of softmax
// in attention pass B.
// ---------------------------------------------------------------------------

typedef _Float16 f16;
typedef __attribute__((ext_vector_type(8)))  _Float16 v8h;
typedef __attribute__((ext_vector_type(16))) _Float16 v16h;
typedef __attribute__((ext_vector_type(8)))  float    v8f;

#define B_  2
#define S_  2048
#define H_  1024
#define NH  16
#define HD  64

// D = A*B + C, 16x16x32 f16 -> f32
__device__ __forceinline__ v8f wmma32(v16h a, v16h b, v8f c) {
  return __builtin_amdgcn_wmma_f32_16x16x32_f16(false, a, false, b, (short)0, c,
                                                false, false);
}

// A fragment, 16(M) x 32(K), from row-major [M x K] source, base at (row0,k0).
// Contiguous-k layout used here (B matches): lane M=lane&15, g=lane>>4; element e -> K=g*16+e.
__device__ __forceinline__ v16h load_a_frag(const f16* base, int ld) {
  int lane = threadIdx.x & 31;
  int M = lane & 15, g = lane >> 4;
  const f16* p = base + (size_t)M * ld;
  return *(const v16h*)(p + g * 16);
}

// B fragment, 32(K) x 16(N), from a row-major [N x K] source. base at (n0,k0).
// Contiguous-k layout used here (A matches): lane N=lane&15, g=lane>>4; element e -> K=g*16+e.
__device__ __forceinline__ v16h load_b_frag_nk(const f16* base, int ld) {
  int lane = threadIdx.x & 31;
  int N = lane & 15, g = lane >> 4;
  return *(const v16h*)(base + (size_t)N * ld + g * 16);
}

// ------------------------------ f32 -> f16 ---------------------------------
__global__ void cvt_f32_f16(const float* __restrict__ src, f16* __restrict__ dst,
                            int n) {
  int i = blockIdx.x * blockDim.x + threadIdx.x;
  if (i < n) dst[i] = (f16)src[i];
}

// ---------------- GEMM core: 32(M) x 64(N) tile per wave, K=1024 -----------
// Explicit ping-pong double buffering: two fragment sets alternate roles per
// 64-wide k iteration, so no rotation copies are required.
template <typename STORE>
__device__ __forceinline__ void gemm_tile(const f16* __restrict__ A,
                                          const f16* __restrict__ W,
                                          int row0, int col0, STORE&& store) {
  const f16* Ar0 = A + (size_t)row0 * H_;
  const f16* Ar1 = A + (size_t)(row0 + 16) * H_;
  const f16* Wc  = W + (size_t)col0 * H_;
  int lane = threadIdx.x & 31;

  v8f acc[2][4] = {};
  v16h aA[2], aB[2], bf[2][4];

  // stage 0 preload (k = 0)
  aA[0] = load_a_frag(Ar0, H_);
  aB[0] = load_a_frag(Ar1, H_);
  #pragma unroll
  for (int nd = 0; nd < 4; ++nd)
    bf[0][nd] = load_b_frag_nk(Wc + (size_t)(nd * 16) * H_, H_);

  #pragma unroll 1
  for (int k0 = 0; k0 < H_; k0 += 64) {
    int k1 = k0 + 32;
    int k2 = (k0 + 64 < H_) ? k0 + 64 : 0;   // clamped dummy on final iteration
    int kp = (k0 + 256 < H_) ? k0 + 256 : k0;
    __builtin_prefetch(Ar0 + (size_t)(lane & 15) * H_ + kp, 0, 1);
    __builtin_prefetch(Wc + (size_t)lane * H_ + kp, 0, 1);
    __builtin_prefetch(Wc + (size_t)(lane + 32) * H_ + kp, 0, 1);

    // stage-1 loads (k1) while stage-0 computes
    aA[1] = load_a_frag(Ar0 + k1, H_);
    aB[1] = load_a_frag(Ar1 + k1, H_);
    #pragma unroll
    for (int nd = 0; nd < 4; ++nd)
      bf[1][nd] = load_b_frag_nk(Wc + (size_t)(nd * 16) * H_ + k1, H_);

    #pragma unroll
    for (int nd = 0; nd < 4; ++nd) {
      acc[0][nd] = wmma32(aA[0], bf[0][nd], acc[0][nd]);
      acc[1][nd] = wmma32(aB[0], bf[0][nd], acc[1][nd]);
    }

    // stage-0 loads (k2, next iteration) while stage-1 computes
    aA[0] = load_a_frag(Ar0 + k2, H_);
    aB[0] = load_a_frag(Ar1 + k2, H_);
    #pragma unroll
    for (int nd = 0; nd < 4; ++nd)
      bf[0][nd] = load_b_frag_nk(Wc + (size_t)(nd * 16) * H_ + k2, H_);

    #pragma unroll
    for (int nd = 0; nd < 4; ++nd) {
      acc[0][nd] = wmma32(aA[1], bf[1][nd], acc[0][nd]);
      acc[1][nd] = wmma32(aB[1], bf[1][nd], acc[1][nd]);
    }
  }

  int N = lane & 15, g = lane >> 4;
  #pragma unroll
  for (int mi = 0; mi < 2; ++mi)
    #pragma unroll
    for (int nd = 0; nd < 4; ++nd)
      #pragma unroll
      for (int v = 0; v < 8; ++v)
        store(row0 + mi * 16 + v + 8 * g, col0 + nd * 16 + N, acc[mi][nd][v]);
}

// -------------------- projection GEMM: Y = X @ W^T (f16 out) ---------------
// vmode==0: row-major (B,S,H). vmode==1: V path -> per-head transposed
//           Vt[b][n][d][j] so PV B-fragments are contiguous.
__global__ __launch_bounds__(256) void gemm_proj(const f16* __restrict__ A,
                                                 const f16* __restrict__ W,
                                                 f16* __restrict__ dst, int vmode) {
  int wid = (int)((blockIdx.x * blockDim.x + threadIdx.x) >> 5);
  int row0 = (wid >> 4) * 32;   // 128 M-tiles
  int col0 = (wid & 15) * 64;   // 16  N-tiles
  gemm_tile(A, W, row0, col0, [&](int r, int c, float val) {
    if (vmode == 0) {
      dst[(size_t)r * H_ + c] = (f16)val;
    } else {
      // raw-reshape head view of per-batch flat offset o = s*H + c
      int b = r >> 11, s = r & (S_ - 1);
      size_t o = (size_t)s * H_ + c;
      int n = (int)(o >> 17);                         // / (S_*HD)
      size_t rem = o & ((size_t)S_ * HD - 1);
      int j = (int)(rem >> 6);
      int d = (int)(rem & 63);
      dst[(size_t)b * NH * HD * S_ + (size_t)n * HD * S_ + (size_t)d * S_ + j] =
          (f16)val;
    }
  });
}

// ---------------------- output GEMM: out = X @ Wo^T (f32 out) --------------
__global__ __launch_bounds__(256) void gemm_out(const f16* __restrict__ A,
                                                const f16* __restrict__ W,
                                                float* __restrict__ dst) {
  int wid = (int)((blockIdx.x * blockDim.x + threadIdx.x) >> 5);
  int row0 = (wid >> 4) * 32;
  int col0 = (wid & 15) * 64;
  gemm_tile(A, W, row0, col0, [&](int r, int c, float val) {
    dst[(size_t)r * H_ + c] = val;
  });
}

// -------------------------- fused attention --------------------------------
// One workgroup per (b, h, 32-row i block). 8 waves split the j space.
// Pass A: scores via WMMA -> LDS -> online row max/sum.
// Pass B: recompute scores, normalize, write f32 attn to d_out, f16 P to LDS,
//         accumulate P@V via WMMA; cross-wave reduce X through LDS atomics.
__global__ __launch_bounds__(256) void attn_kernel(const f16* __restrict__ Qp,
                                                   const f16* __restrict__ Kp,
                                                   const f16* __restrict__ Vt,
                                                   f16* __restrict__ Xh,
                                                   float* __restrict__ attn_out) {
  __shared__ float Sbuf[8][32][32];   // per-wave score staging
  __shared__ f16   Pbuf[8][32][32];   // per-wave f16 probabilities
  __shared__ float Mw[8][32], Zw[8][32];
  __shared__ float Xred[32][64];

  int bid  = blockIdx.x;              // b*NH*64 + h*64 + iblk
  int iblk = bid & 63;
  int h    = (bid >> 6) & (NH - 1);
  int b    = bid >> 10;
  int i0   = iblk * 32;

  int tid = threadIdx.x;
  int w = tid >> 5, lane = tid & 31;
  int N = lane & 15, g = lane >> 4;

  for (int t = tid; t < 32 * 64; t += 256) ((float*)Xred)[t] = 0.f;

  // raw-reshape head bases: contiguous 2048x64, ld = HD
  const f16* Qb = Qp + (size_t)b * S_ * H_ + (size_t)h * S_ * HD + (size_t)i0 * HD;
  const f16* Kb = Kp + (size_t)b * S_ * H_ + (size_t)h * S_ * HD;
  const f16* Vb = Vt + (size_t)b * NH * HD * S_ + (size_t)h * HD * S_;  // [d][j]

  // Q fragments are invariant across chunks and both passes: load once.
  v16h qf[2][2];
  #pragma unroll
  for (int mi = 0; mi < 2; ++mi)
    #pragma unroll
    for (int dk = 0; dk < 2; ++dk)
      qf[mi][dk] = load_a_frag(Qb + (size_t)(mi * 16) * HD + dk * 32, HD);

  const float scale = 0.125f;  // 1/sqrt(64)
  float m = -1e30f, z = 0.f;

  // ---------------- pass A: online softmax statistics ----------------
  for (int c = 0; c < 8; ++c) {
    int j0 = (c * 8 + w) * 32;
    // prefetch next chunk's K stripe (32 rows x 128B, contiguous)
    int j0n = j0 + 8 * 32;
    if (c < 7) __builtin_prefetch(Kb + (size_t)(j0n + lane) * HD, 0, 1);
    #pragma unroll
    for (int mi = 0; mi < 2; ++mi)
      #pragma unroll
      for (int jj = 0; jj < 2; ++jj) {
        v16h bk0 = load_b_frag_nk(Kb + (size_t)(j0 + jj * 16) * HD, HD);
        v16h bk1 = load_b_frag_nk(Kb + (size_t)(j0 + jj * 16) * HD + 32, HD);
        v8f sc = {};
        sc = wmma32(qf[mi][0], bk0, sc);
        sc = wmma32(qf[mi][1], bk1, sc);
        #pragma unroll
        for (int v = 0; v < 8; ++v)
          Sbuf[w][mi * 16 + v + 8 * g][jj * 16 + N] = sc[v];
      }
    __syncthreads();
    float mc = -1e30f;
    for (int t = 0; t < 32; ++t) mc = fmaxf(mc, Sbuf[w][lane][t] * scale);
    float mn = fmaxf(m, mc);
    float zc = 0.f;
    for (int t = 0; t < 32; ++t) zc += __expf(Sbuf[w][lane][t] * scale - mn);
    z = z * __expf(m - mn) + zc;
    m = mn;
    __syncthreads();
  }
  Mw[w][lane] = m;
  Zw[w][lane] = z;
  __syncthreads();
  float gM = -1e30f;
  for (int ww = 0; ww < 8; ++ww) gM = fmaxf(gM, Mw[ww][lane]);
  float gZ = 0.f;
  for (int ww = 0; ww < 8; ++ww) gZ += Zw[ww][lane] * __expf(Mw[ww][lane] - gM);
  float invZ = 1.f / gZ;

  // ---------------- pass B: attn weights + P@V ----------------
  v8f xacc[2][4] = {};
  float* attn_base = attn_out + (((size_t)(b * NH + h)) * S_ + i0) * S_;

  for (int c = 0; c < 8; ++c) {
    int j0 = (c * 8 + w) * 32;
    int j0n = j0 + 8 * 32;
    if (c < 7) {
      // prefetch next chunk's K stripe and V columns
      __builtin_prefetch(Kb + (size_t)(j0n + lane) * HD, 0, 1);
      __builtin_prefetch(Vb + (size_t)lane * S_ + j0n, 0, 1);
      __builtin_prefetch(Vb + (size_t)(lane + 32) * S_ + j0n, 0, 1);
    }
    #pragma unroll
    for (int mi = 0; mi < 2; ++mi)
      #pragma unroll
      for (int jj = 0; jj < 2; ++jj) {
        v16h bk0 = load_b_frag_nk(Kb + (size_t)(j0 + jj * 16) * HD, HD);
        v16h bk1 = load_b_frag_nk(Kb + (size_t)(j0 + jj * 16) * HD + 32, HD);
        v8f sc = {};
        sc = wmma32(qf[mi][0], bk0, sc);
        sc = wmma32(qf[mi][1], bk1, sc);
        #pragma unroll
        for (int v = 0; v < 8; ++v)
          Sbuf[w][mi * 16 + v + 8 * g][jj * 16 + N] = sc[v];
      }
    // hoist V fragment loads: in flight during softmax/exp below
    v16h vfrag[4];
    #pragma unroll
    for (int nd = 0; nd < 4; ++nd)
      vfrag[nd] = load_b_frag_nk(Vb + (size_t)(nd * 16) * S_ + j0, S_);
    __syncthreads();
    for (int t = 0; t < 32; ++t) {
      float p = __expf(Sbuf[w][lane][t] * scale - gM) * invZ;
      attn_base[(size_t)lane * S_ + j0 + t] = p;
      Pbuf[w][lane][t] = (f16)p;
    }
    __syncthreads();
    #pragma unroll
    for (int mi = 0; mi < 2; ++mi) {
      v16h a = load_a_frag(&Pbuf[w][mi * 16][0], 32);
      #pragma unroll
      for (int nd = 0; nd < 4; ++nd)
        xacc[mi][nd] = wmma32(a, vfrag[nd], xacc[mi][nd]);
    }
    __syncthreads();
  }

  // cross-wave reduction of X (ds_add_f32)
  #pragma unroll
  for (int mi = 0; mi < 2; ++mi)
    #pragma unroll
    for (int nd = 0; nd < 4; ++nd)
      #pragma unroll
      for (int v = 0; v < 8; ++v)
        atomicAdd(&Xred[mi * 16 + v + 8 * g][nd * 16 + N], xacc[mi][nd][v]);
  __syncthreads();

  // write X tile as f16, layout (b, i, n*d)
  for (int t = tid; t < 32 * 64; t += 256) {
    int r = t >> 6, d = t & 63;
    Xh[((size_t)b * S_ + i0 + r) * H_ + h * HD + d] = (f16)Xred[r][d];
  }
}

// ---------------------------------------------------------------------------
extern "C" void kernel_launch(void* const* d_in, const int* in_sizes, int n_in,
                              void* d_out, int out_size, void* d_ws, size_t ws_size,
                              hipStream_t stream) {
  (void)in_sizes; (void)n_in; (void)out_size; (void)ws_size;
  const float* q  = (const float*)d_in[0];
  const float* k  = (const float*)d_in[1];
  const float* v  = (const float*)d_in[2];
  const float* Wq = (const float*)d_in[3];
  const float* Wk = (const float*)d_in[4];
  const float* Wv = (const float*)d_in[5];
  const float* Wo = (const float*)d_in[6];

  float* out  = (float*)d_out;                       // (B,S,H) f32
  float* attn = out + (size_t)B_ * S_ * H_;          // (B,NH,S,S) f32

  const size_t E  = (size_t)B_ * S_ * H_;            // 4,194,304
  const size_t WE = (size_t)H_ * H_;                 // 1,048,576
  f16* ws = (f16*)d_ws;
  f16* qf = ws;           f16* kf = qf + E;  f16* vf = kf + E;
  f16* wq = vf + E;       f16* wk = wq + WE; f16* wv = wk + WE; f16* wo = wv + WE;
  f16* Qp = wo + WE;      f16* Kp = Qp + E;
  f16* Vt = Kp + E;       f16* Xh = Vt + E;          // total 32M halves = 64 MB

  // f32 -> f16 conversions
  cvt_f32_f16<<<(int)((E  + 255) / 256), 256, 0, stream>>>(q,  qf, (int)E);
  cvt_f32_f16<<<(int)((E  + 255) / 256), 256, 0, stream>>>(k,  kf, (int)E);
  cvt_f32_f16<<<(int)((E  + 255) / 256), 256, 0, stream>>>(v,  vf, (int)E);
  cvt_f32_f16<<<(int)((WE + 255) / 256), 256, 0, stream>>>(Wq, wq, (int)WE);
  cvt_f32_f16<<<(int)((WE + 255) / 256), 256, 0, stream>>>(Wk, wk, (int)WE);
  cvt_f32_f16<<<(int)((WE + 255) / 256), 256, 0, stream>>>(Wv, wv, (int)WE);
  cvt_f32_f16<<<(int)((WE + 255) / 256), 256, 0, stream>>>(Wo, wo, (int)WE);

  // projections: 128 x 16 = 2048 wave-tiles of 32x64 => 256 blocks of 8 waves
  gemm_proj<<<256, 256, 0, stream>>>(qf, wq, Qp, 0);
  gemm_proj<<<256, 256, 0, stream>>>(kf, wk, Kp, 0);
  gemm_proj<<<256, 256, 0, stream>>>(vf, wv, Vt, 1);

  // fused attention: B*NH*(S/32) = 2048 workgroups
  attn_kernel<<<2048, 256, 0, stream>>>(Qp, Kp, Vt, Xh, attn);

  // output projection
  gemm_out<<<256, 256, 0, stream>>>(Xh, wo, out);
}
